// MultiHeadAttention_45689862095174
// MI455X (gfx1250) — matrix operand order repaired and checked by reading, not compiled, not run on hardware
//
#include <hip/hip_runtime.h>

#define TT 2048
#define DD 1024
#define HH 16
#define DH 64
#define BB 4

// 0.125 (Dh^-0.5) * log2(e), folded into Q at bf16-write time
#define ATT_SC 0.180336880111f

typedef __attribute__((ext_vector_type(16))) __bf16 v16bf;
typedef __attribute__((ext_vector_type(8)))  float  v8f;

union FragB16 { unsigned int u[8]; v16bf v; };

__device__ __forceinline__ unsigned short f2bf(float f) {
  unsigned int u = __builtin_bit_cast(unsigned int, f);
  u += 0x7FFFu + ((u >> 16) & 1u);           // round-to-nearest-even
  return (unsigned short)(u >> 16);
}
__device__ __forceinline__ unsigned int pack2(float lo, float hi) {
  return (unsigned int)f2bf(lo) | ((unsigned int)f2bf(hi) << 16);
}
__device__ __forceinline__ v8f wmma_bf16(const FragB16& a, const FragB16& b, v8f c) {
  return __builtin_amdgcn_wmma_f32_16x16x32_bf16(false, a.v, false, b.v,
                                                 (short)0, c, false, false);
}

// DPP16 lane permute (stays within 16-lane rows; wave32-safe)
template<int CTRL>
__device__ __forceinline__ float dpp_perm(float v) {
  return __builtin_bit_cast(float,
      __builtin_amdgcn_update_dpp(0, __builtin_bit_cast(int, v), CTRL, 0xF, 0xF, true));
}
// Butterfly over 16-lane group via xor masks {1,2,7,15} (span = full 4-bit group)
__device__ __forceinline__ float rmax16(float v) {
  v = fmaxf(v, dpp_perm<0xB1>(v));   // quad_perm [1,0,3,2] = xor1
  v = fmaxf(v, dpp_perm<0x4E>(v));   // quad_perm [2,3,0,1] = xor2
  v = fmaxf(v, dpp_perm<0x141>(v));  // row_half_mirror     = xor7
  v = fmaxf(v, dpp_perm<0x140>(v));  // row_mirror          = xor15
  return v;
}
__device__ __forceinline__ float rsum16(float v) {
  v += dpp_perm<0xB1>(v);
  v += dpp_perm<0x4E>(v);
  v += dpp_perm<0x141>(v);
  v += dpp_perm<0x140>(v);
  return v;
}

// MODE 0: A = fp32 x, B = fp32 w_qkv (ncols=3D), epilogue scatters bf16 Q*sc/K/V^T
// MODE 1: A = bf16 O, B = fp32 w_proj (ncols=D), epilogue adds bias, fp32 out
template<int MODE>
__global__ __launch_bounds__(256) void gemm_kernel(
    const float* __restrict__ Af, const unsigned short* __restrict__ Ah,
    const float* __restrict__ W, int ncols,
    unsigned short* __restrict__ Qo, unsigned short* __restrict__ Ko,
    unsigned short* __restrict__ Vto,
    const float* __restrict__ bias, float* __restrict__ Co)
{
  __shared__ unsigned int As[128][20];  // stride 80B: 16B-aligned rows, conflict-free
  __shared__ unsigned int Bs[64][20];   // [n][k-pair], B stored transposed
  const int tid  = threadIdx.x;
  const int lane = tid & 31, wave = tid >> 5;
  const int wm = wave >> 1, wn = wave & 1;      // 4x2 wave grid -> 128x64 block tile
  const int row0 = blockIdx.y * 128;
  const int col0 = blockIdx.x * 64;
  const int hq = lane >> 4, l16 = lane & 15;

  v8f acc[2][2] = {};

  for (int k0 = 0; k0 < DD; k0 += 32) {
    if (k0 + 32 < DD) {                 // prefetch next k-step tiles
      if constexpr (MODE == 0)
        __builtin_prefetch(Af + (size_t)(row0 + (tid >> 3)) * DD + k0 + 32 + (tid & 7) * 4, 0, 3);
      else
        __builtin_prefetch(Ah + (size_t)(row0 + (tid >> 2)) * DD + k0 + 32, 0, 3);
      __builtin_prefetch(W + (size_t)(k0 + 32 + (tid >> 5)) * ncols + col0 + (tid & 31) * 2, 0, 3);
    }
    if constexpr (MODE == 0) {
      const int r = tid >> 3, c = (tid & 7) * 4;
#pragma unroll
      for (int p = 0; p < 4; ++p) {
        const float4 f = *(const float4*)(Af + (size_t)(row0 + p * 32 + r) * DD + k0 + c);
        As[p * 32 + r][(c >> 1)]     = pack2(f.x, f.y);
        As[p * 32 + r][(c >> 1) + 1] = pack2(f.z, f.w);
      }
    } else {
      const int r = tid >> 2, c4 = (tid & 3) * 4;
#pragma unroll
      for (int p = 0; p < 2; ++p) {
        const uint4 q = *(const uint4*)(Ah + (size_t)(row0 + p * 64 + r) * DD + k0 + c4 * 2);
        *(uint4*)&As[p * 64 + r][c4] = q;
      }
    }
    {
      const int kk = tid >> 5, n2 = (tid & 31) * 2;
      unsigned short* bsh = (unsigned short*)&Bs[0][0];   // row stride 40 shorts
#pragma unroll
      for (int p = 0; p < 4; ++p) {
        const int kr = p * 8 + kk;
        const float2 f = *(const float2*)(W + (size_t)(k0 + kr) * ncols + col0 + n2);
        bsh[(size_t)n2 * 40 + kr]       = f2bf(f.x);
        bsh[(size_t)(n2 + 1) * 40 + kr] = f2bf(f.y);
      }
    }
    __syncthreads();

    FragB16 a[2], b[2];
#pragma unroll
    for (int mt = 0; mt < 2; ++mt) {    // A-frag: 4 x ds_load_b64
      const unsigned int* ap = &As[wm * 32 + mt * 16 + l16][4 * hq];
      const uint2 a0 = *(const uint2*)(ap + 0);
      const uint2 a1 = *(const uint2*)(ap + 2);
      const uint2 a2 = *(const uint2*)(ap + 8);
      const uint2 a3 = *(const uint2*)(ap + 10);
      a[mt].u[0] = a0.x; a[mt].u[1] = a0.y; a[mt].u[2] = a1.x; a[mt].u[3] = a1.y;
      a[mt].u[4] = a2.x; a[mt].u[5] = a2.y; a[mt].u[6] = a3.x; a[mt].u[7] = a3.y;
    }
#pragma unroll
    for (int nt = 0; nt < 2; ++nt) {    // B-frag: 2 x ds_load_b128
      const uint4* bp = (const uint4*)&Bs[wn * 32 + nt * 16 + l16][4 * hq];
      const uint4 b0 = bp[0], b1 = bp[2];
      b[nt].u[0] = b0.x; b[nt].u[1] = b0.y; b[nt].u[2] = b0.z; b[nt].u[3] = b0.w;
      b[nt].u[4] = b1.x; b[nt].u[5] = b1.y; b[nt].u[6] = b1.z; b[nt].u[7] = b1.w;
    }
#pragma unroll
    for (int mt = 0; mt < 2; ++mt)
#pragma unroll
      for (int nt = 0; nt < 2; ++nt)
        acc[mt][nt] = wmma_bf16(a[mt], b[nt], acc[mt][nt]);
    __syncthreads();
  }

#pragma unroll
  for (int mt = 0; mt < 2; ++mt) {
#pragma unroll
    for (int nt = 0; nt < 2; ++nt) {
      const int jcol = col0 + wn * 32 + nt * 16 + l16;
      if constexpr (MODE == 0) {
        const int which = jcol >> 10;          // 0=Q 1=K 2=V
        const int dcol = jcol & (DD - 1);
        const int hh = dcol >> 6, dh = dcol & (DH - 1);
        const float qsc = (which == 0) ? ATT_SC : 1.0f;  // fold softmax scale into Q
#pragma unroll
        for (int r = 0; r < 8; ++r) {
          const int g = row0 + wm * 32 + mt * 16 + r + 8 * hq;
          const int bb = g >> 11, t = g & (TT - 1);
          const unsigned short vv = f2bf(acc[mt][nt][r] * qsc);
          const size_t bh = (size_t)bb * HH + hh;
          if (which == 0)      Qo [(bh * TT + t) * DH + dh] = vv;
          else if (which == 1) Ko [(bh * TT + t) * DH + dh] = vv;
          else                 Vto[(bh * DH + dh) * TT + t] = vv;   // V transposed
        }
      } else {
        const float bj = bias[jcol];
#pragma unroll
        for (int r = 0; r < 8; ++r) {
          const int g = row0 + wm * 32 + mt * 16 + r + 8 * hq;
          Co[(size_t)g * DD + jcol] = acc[mt][nt][r] + bj;
        }
      }
    }
  }
}

// One 32-key chunk of online-softmax attention. MASK=true only on diagonal chunks.
template<bool MASK>
__device__ __forceinline__ void attn_chunk(
    const int k0, const int q0, const int hq, const int l16,
    const unsigned short* __restrict__ Kp, const unsigned short* __restrict__ Vp,
    const FragB16* aq, unsigned int* __restrict__ plw,   // wave LDS, 20 uints/row
    v8f* o, float* mrow, float* lrow)
{
  v8f st[2] = {};
#pragma unroll
  for (int nt = 0; nt < 2; ++nt) {
#pragma unroll
    for (int s = 0; s < 2; ++s) {
      FragB16 bk;  // B-frag: lane = key column, two 16B pieces of a K row (k 8h.., 16+8h..)
      const unsigned short* kr = Kp + (size_t)(k0 + nt * 16 + l16) * DH + s * 32 + 8 * hq;
      const uint4 lo = *(const uint4*)kr;
      const uint4 hi = *(const uint4*)(kr + 16);
      bk.u[0] = lo.x; bk.u[1] = lo.y; bk.u[2] = lo.z; bk.u[3] = lo.w;
      bk.u[4] = hi.x; bk.u[5] = hi.y; bk.u[6] = hi.z; bk.u[7] = hi.w;
      st[nt] = wmma_bf16(aq[s], bk, st[nt]);
    }
  }
  if (!MASK) {   // prefetch next key chunk (bulk loop only)
    __builtin_prefetch(Kp + (size_t)(k0 + 32 + l16) * DH + 16 * hq, 0, 3);
    __builtin_prefetch(Vp + (size_t)l16 * TT + k0 + 32 + 16 * hq, 0, 3);
  }
#pragma unroll
  for (int r = 0; r < 8; ++r) {
    float v0 = st[0][r];       // already scaled: Q carries Dh^-0.5*log2(e)
    float v1 = st[1][r];
    if (MASK) {
      const int qi = q0 + r + 8 * hq;
      if (k0 + l16 > qi)      v0 = -1e30f;
      if (k0 + 16 + l16 > qi) v1 = -1e30f;
    }
    const float cm    = rmax16(fmaxf(v0, v1));
    const float mnew  = fmaxf(mrow[r], cm);
    const float alpha = __builtin_amdgcn_exp2f(mrow[r] - mnew);
    const float p0 = __builtin_amdgcn_exp2f(v0 - mnew);
    const float p1 = __builtin_amdgcn_exp2f(v1 - mnew);
    lrow[r] = lrow[r] * alpha + rsum16(p0 + p1);
    mrow[r] = mnew;
    o[0][r] *= alpha; o[1][r] *= alpha; o[2][r] *= alpha; o[3][r] *= alpha;
    unsigned short* pls = (unsigned short*)(plw + (size_t)(r + 8 * hq) * 20);
    pls[l16]      = f2bf(p0);
    pls[16 + l16] = f2bf(p1);
  }
  // P (16x32 bf16) as A-fragment via LDS round trip: 4 x ds_load_b64
  FragB16 ap;
  const unsigned int* pu = plw + (size_t)l16 * 20 + 4 * hq;
  const uint2 a0 = *(const uint2*)(pu + 0);
  const uint2 a1 = *(const uint2*)(pu + 2);
  const uint2 a2 = *(const uint2*)(pu + 8);
  const uint2 a3 = *(const uint2*)(pu + 10);
  ap.u[0] = a0.x; ap.u[1] = a0.y; ap.u[2] = a1.x; ap.u[3] = a1.y;
  ap.u[4] = a2.x; ap.u[5] = a2.y; ap.u[6] = a3.x; ap.u[7] = a3.y;
#pragma unroll
  for (int n = 0; n < 4; ++n) {
    FragB16 bv;  // B-frag from V^T: lane = Dh column, two 16B pieces over keys (8h.., 16+8h..)
    const unsigned short* vr = Vp + (size_t)(n * 16 + l16) * TT + k0 + 8 * hq;
    const uint4 lo = *(const uint4*)vr;
    const uint4 hi = *(const uint4*)(vr + 16);
    bv.u[0] = lo.x; bv.u[1] = lo.y; bv.u[2] = lo.z; bv.u[3] = lo.w;
    bv.u[4] = hi.x; bv.u[5] = hi.y; bv.u[6] = hi.z; bv.u[7] = hi.w;
    o[n] = wmma_bf16(ap, bv, o[n]);
  }
}

// Flash attention: 1 wave per 16-query tile, online softmax, causal.
__global__ __launch_bounds__(128) void attn_kernel(
    const unsigned short* __restrict__ Q, const unsigned short* __restrict__ K,
    const unsigned short* __restrict__ Vt, unsigned short* __restrict__ O)
{
  __shared__ unsigned int Pl[4][16][20];  // per-wave P staging, 80B row stride
  const int wave = threadIdx.x >> 5, lane = threadIdx.x & 31;
  const int task = blockIdx.x * 4 + wave;
  const int q0 = (task & (TT / 16 - 1)) * 16;
  const int bh = task >> 7;                       // T/16 == 128
  const unsigned short* Qp = Q  + (size_t)bh * TT * DH;
  const unsigned short* Kp = K  + (size_t)bh * TT * DH;
  const unsigned short* Vp = Vt + (size_t)bh * DH * TT;
  const int hq = lane >> 4, l16 = lane & 15;
  unsigned int* plw = &Pl[wave][0][0];

  // Q A-fragments for both 32-wide k-steps over Dh=64; kept in VGPRs all kernel
  FragB16 aq[2];
#pragma unroll
  for (int s = 0; s < 2; ++s) {
    const unsigned short* qr = Qp + (size_t)(q0 + l16) * DH + s * 32 + 8 * hq;
    const uint4 lo = *(const uint4*)qr;
    const uint4 hi = *(const uint4*)(qr + 16);
    aq[s].u[0] = lo.x; aq[s].u[1] = lo.y; aq[s].u[2] = lo.z; aq[s].u[3] = lo.w;
    aq[s].u[4] = hi.x; aq[s].u[5] = hi.y; aq[s].u[6] = hi.z; aq[s].u[7] = hi.w;
  }

  v8f o[4] = {};
  float mrow[8], lrow[8];
#pragma unroll
  for (int r = 0; r < 8; ++r) { mrow[r] = -1e30f; lrow[r] = 0.0f; }

  const int nfull   = q0 >> 5;                   // chunks fully below the diagonal
  const int nchunks = (q0 + 16 + 31) >> 5;
  for (int kb = 0; kb < nfull; ++kb)
    attn_chunk<false>(kb * 32, q0, hq, l16, Kp, Vp, aq, plw, o, mrow, lrow);
  for (int kb = nfull; kb < nchunks; ++kb)
    attn_chunk<true>(kb * 32, q0, hq, l16, Kp, Vp, aq, plw, o, mrow, lrow);

  const int bb = bh >> 4, hh = bh & (HH - 1);
#pragma unroll
  for (int r = 0; r < 8; ++r) {
    const float inv = __builtin_amdgcn_rcpf(lrow[r]);
    const int t = q0 + r + 8 * hq;
    unsigned short* orow = O + (size_t)(bb * TT + t) * DD + hh * DH;
#pragma unroll
    for (int n = 0; n < 4; ++n) orow[n * 16 + l16] = f2bf(o[n][r] * inv);
  }
}

extern "C" void kernel_launch(void* const* d_in, const int* in_sizes, int n_in,
                              void* d_out, int out_size, void* d_ws, size_t ws_size,
                              hipStream_t stream) {
  (void)in_sizes; (void)n_in; (void)out_size; (void)ws_size;
  const float* x      = (const float*)d_in[0];
  const float* w_qkv  = (const float*)d_in[1];
  const float* w_proj = (const float*)d_in[2];
  const float* b_proj = (const float*)d_in[3];
  float* out = (float*)d_out;

  const size_t elems = (size_t)BB * TT * DD;     // 8,388,608
  unsigned short* Q  = (unsigned short*)d_ws;    // (B,H,T,Dh) bf16 * sc  16 MB
  unsigned short* K  = Q + elems;                // (B,H,T,Dh) bf16       16 MB
  unsigned short* Vt = K + elems;                // (B,H,Dh,T) bf16       16 MB
  unsigned short* O  = Vt + elems;               // (B,T,D)    bf16       16 MB

  dim3 g1((3 * DD) / 64, (BB * TT) / 128);       // 48 x 64
  gemm_kernel<0><<<g1, 256, 0, stream>>>(x, nullptr, w_qkv, 3 * DD,
                                         Q, K, Vt, nullptr, nullptr);

  attn_kernel<<<(BB * HH * (TT / 16)) / 4, 128, 0, stream>>>(Q, K, Vt, O);

  dim3 g3(DD / 64, (BB * TT) / 128);             // 16 x 64
  gemm_kernel<1><<<g3, 256, 0, stream>>>(nullptr, O, w_proj, DD,
                                         nullptr, nullptr, nullptr, b_proj, out);
}
